// MyGAT_70042326663941
// MI455X (gfx1250) — matrix operand order repaired and checked by reading, not compiled, not run on hardware
//
#include <hip/hip_runtime.h>
#include <hip/hip_bf16.h>

typedef _Float16 h8   __attribute__((ext_vector_type(8)));
typedef _Float16 v16h __attribute__((ext_vector_type(16)));
typedef float    v8f  __attribute__((ext_vector_type(8)));

#define NEG_SLOPE 0.2f

// ---------- helpers ----------
__device__ __forceinline__ unsigned enc_key(float x) {
    unsigned u = __float_as_uint(x);
    return ((int)u < 0) ? ~u : (u | 0x80000000u);
}
__device__ __forceinline__ float dec_key(unsigned k) {
    unsigned u = (k & 0x80000000u) ? (k & 0x7FFFFFFFu) : ~k;
    return __uint_as_float(u);
}

// ---------- conversion / transpose ----------
__global__ void k_cvt_f16(const float* __restrict__ src, _Float16* __restrict__ dst, int n) {
    int i = blockIdx.x * blockDim.x + threadIdx.x;
    if (i < n) dst[i] = (_Float16)src[i];
}

// W: [K][Ncol] f32 row-major  ->  WT: [Ncol][K] f16
__global__ void k_transpose_f16(const float* __restrict__ W, _Float16* __restrict__ WT,
                                int K, int Ncol) {
    int i = blockIdx.x * blockDim.x + threadIdx.x;
    if (i >= K * Ncol) return;
    int k = i / Ncol, c = i % Ncol;
    WT[(size_t)c * K + k] = (_Float16)W[i];
}

// ---------- WMMA GEMM: C[M][Ncol] = A[M][K] * B[K][Ncol] (B given as BT[Ncol][K]) ----------
__global__ void k_gemm_wmma(const _Float16* __restrict__ A, const _Float16* __restrict__ BT,
                            float* __restrict__ C, int Mtiles, int K, int Ncol) {
    const int wave = threadIdx.x >> 5;
    const int lane = threadIdx.x & 31;
    const int tM = blockIdx.y * (blockDim.x >> 5) + wave;
    const int tN = blockIdx.x;
    if (tM >= Mtiles) return;                       // wave-uniform guard: EXEC stays all-1s

    const int l16  = lane & 15;
    const int half = lane >> 4;

    // A fragment source: row (tM*16 + l16), two 8-half runs at K = khalf*8 and 16+khalf*8
    const _Float16* arow = A  + (size_t)(tM * 16 + l16) * K + half * 8;
    // B fragment source: WT row (tN*16 + l16), two 8-half runs at K = khalf*8 and 16+khalf*8
    const _Float16* brow = BT + (size_t)(tN * 16 + l16) * K + half * 8;

    v8f acc = {};
    for (int k0 = 0; k0 < K; k0 += 32) {
        h8 alo = *(const h8*)(arow + k0);
        h8 ahi = *(const h8*)(arow + k0 + 16);
        h8 blo = *(const h8*)(brow + k0);
        h8 bhi = *(const h8*)(brow + k0 + 16);
        if (k0 + 32 < K) {
            __builtin_prefetch(arow + k0 + 32, 0, 1);
            __builtin_prefetch(brow + k0 + 32, 0, 1);
        }
        v16h a = __builtin_shufflevector(alo, ahi, 0,1,2,3,4,5,6,7,8,9,10,11,12,13,14,15);
        v16h b = __builtin_shufflevector(blo, bhi, 0,1,2,3,4,5,6,7,8,9,10,11,12,13,14,15);
        acc = __builtin_amdgcn_wmma_f32_16x16x32_f16(false, a, false, b,
                                                     (short)0, acc, false, false);
    }

    // C/D layout: VGPR r -> row r + 8*half, col l16
    float* crow = C + (size_t)(tM * 16) * Ncol + tN * 16 + l16;
    const int mbase = half * 8;
#pragma unroll
    for (int r = 0; r < 8; ++r)
        crow[(size_t)(r + mbase) * Ncol] = acc[r];
}

// scalar tail (rows not covered by 16-row tiles; unused for N=50000 but kept for generality)
__global__ void k_gemm_tail(const _Float16* __restrict__ A, const _Float16* __restrict__ BT,
                            float* __restrict__ C, int rowStart, int M, int K, int Ncol) {
    int i = blockIdx.x * blockDim.x + threadIdx.x;
    int rows = M - rowStart;
    if (i >= rows * Ncol) return;
    int r = rowStart + i / Ncol, c = i % Ncol;
    const _Float16* a = A + (size_t)r * K;
    const _Float16* b = BT + (size_t)c * K;
    float acc = 0.f;
    for (int k = 0; k < K; ++k) acc += (float)a[k] * (float)b[k];
    C[(size_t)r * Ncol + c] = acc;
}

// ---------- attention logits: al_s[n,h] = <h[n,h,:], a_src[h,:]> ----------
__global__ void k_attn(const float* __restrict__ h, const float* __restrict__ a_src,
                       const float* __restrict__ a_dst, float* __restrict__ al_s,
                       float* __restrict__ al_d, int N, int H, int Cc) {
    int i = blockIdx.x * blockDim.x + threadIdx.x;
    if (i >= N * H) return;
    int n = i / H, hh = i % H;
    const float* row = h + ((size_t)n * H + hh) * Cc;
    const float* as  = a_src + hh * Cc;
    const float* ad  = a_dst + hh * Cc;
    float ss = 0.f, sd = 0.f;
    for (int c = 0; c < Cc; ++c) { float v = row[c]; ss += v * as[c]; sd += v * ad[c]; }
    al_s[i] = ss; al_d[i] = sd;
}

// ---------- pass 1: leaky-relu logit + segment max (monotone-uint atomicMax) ----------
__global__ void k_edge_max(const int* __restrict__ src, const int* __restrict__ dst,
                           const float* __restrict__ al_s, const float* __restrict__ al_d,
                           float* __restrict__ ebuf, unsigned* __restrict__ mkey,
                           int E, int EE, int H) {
    int i = blockIdx.x * blockDim.x + threadIdx.x;
    if (i >= EE * H) return;
    int hh = i % H, e = i / H;
    int s, d;
    if (e < E) { s = src[e]; d = dst[e]; } else { s = d = e - E; }
    float v = al_s[s * H + hh] + al_d[d * H + hh];
    v = v > 0.f ? v : NEG_SLOPE * v;
    ebuf[i] = v;
    atomicMax(&mkey[d * H + hh], enc_key(v));
}

// ---------- pass 2: exp(e - m[d]) + segment sum ----------
__global__ void k_edge_exp(const int* __restrict__ src, const int* __restrict__ dst,
                           float* __restrict__ ebuf, const unsigned* __restrict__ mkey,
                           float* __restrict__ denom, int E, int EE, int H) {
    int i = blockIdx.x * blockDim.x + threadIdx.x;
    if (i >= EE * H) return;
    int hh = i % H, e = i / H;
    int d = (e < E) ? dst[e] : (e - E);
    float m  = dec_key(mkey[d * H + hh]);
    float ex = __expf(ebuf[i] - m);
    ebuf[i] = ex;
    atomicAdd(&denom[d * H + hh], ex);
}

// ---------- pass 3: agg[d,h,c] += h[s,h,c] * alpha ----------
__global__ void k_scatter(const int* __restrict__ src, const int* __restrict__ dst,
                          const float* __restrict__ h, const float* __restrict__ ebuf,
                          const float* __restrict__ denom, float* __restrict__ agg,
                          int E, int EE, int H, int Cc) {
    int i = blockIdx.x * blockDim.x + threadIdx.x;
    if (i >= EE * H * Cc) return;
    int c = i % Cc;
    int t = i / Cc;
    int hh = t % H;
    int e  = t / H;
    int s, d;
    if (e < E) { s = src[e]; d = dst[e]; } else { s = d = e - E; }
    float alpha = ebuf[t] / denom[d * H + hh];
    atomicAdd(&agg[((size_t)d * H + hh) * Cc + c],
              h[((size_t)s * H + hh) * Cc + c] * alpha);
}

// ---------- bias + ELU; write f32 in place + f16 copy for next GEMM ----------
__global__ void k_finalize(float* __restrict__ agg, const float* __restrict__ bias,
                           _Float16* __restrict__ out16, int total, int HC) {
    int i = blockIdx.x * blockDim.x + threadIdx.x;
    if (i >= total) return;
    float v = agg[i] + bias[i % HC];
    v = v > 0.f ? v : (__expf(v) - 1.0f);
    agg[i] = v;
    out16[i] = (_Float16)v;
}

// ---------- global mean pool + FC ----------
__global__ void k_pool(const float* __restrict__ h, const int* __restrict__ batch,
                       float* __restrict__ sums, float* __restrict__ cnts, int N, int Cc) {
    int i = blockIdx.x * blockDim.x + threadIdx.x;
    if (i >= N * Cc) return;
    int n = i / Cc, c = i % Cc;
    int g = batch[n];
    atomicAdd(&sums[g * Cc + c], h[i]);
    if (c == 0) atomicAdd(&cnts[g], 1.0f);
}

__global__ void k_fc(const float* __restrict__ sums, const float* __restrict__ cnts,
                     const float* __restrict__ fcW, const float* __restrict__ fcb,
                     float* __restrict__ out, int G, int Cc, int O) {
    int i = blockIdx.x * blockDim.x + threadIdx.x;
    if (i >= G * O) return;
    int g = i / O, o = i % O;
    float inv = 1.0f / fmaxf(cnts[g], 1.0f);
    float acc = fcb[o];
    for (int c = 0; c < Cc; ++c) acc += (sums[g * Cc + c] * inv) * fcW[c * O + o];
    out[i] = acc;
}

// ---------- host orchestration ----------
static inline int cdiv(long long a, long long b) { return (int)((a + b - 1) / b); }

struct LayerBufs {
    float *hbuf, *agg, *ebuf, *als, *ald, *denom;
    unsigned *mkey;
};

static void run_gat_layer(hipStream_t stream,
                          const _Float16* act16_in, const _Float16* WT,
                          int K, int H, int Cc,
                          const float* a_src, const float* a_dst, const float* bias,
                          const int* src, const int* dst, int E, int EE, int N,
                          const LayerBufs& B, _Float16* act16_out) {
    const int HC = H * Cc;
    const int Ncol = HC;
    const int Mtiles = N / 16;
    const int TB = 256;

    // GEMM: h = act16_in @ W
    dim3 ggrid(Ncol / 16, cdiv(Mtiles, TB / 32));
    k_gemm_wmma<<<ggrid, TB, 0, stream>>>(act16_in, WT, B.hbuf, Mtiles, K, Ncol);
    int tail = N - Mtiles * 16;
    if (tail > 0)
        k_gemm_tail<<<cdiv((long long)tail * Ncol, TB), TB, 0, stream>>>(
            act16_in, WT, B.hbuf, Mtiles * 16, N, K, Ncol);

    // attention logits
    k_attn<<<cdiv((long long)N * H, TB), TB, 0, stream>>>(B.hbuf, a_src, a_dst,
                                                          B.als, B.ald, N, H, Cc);
    // zero segment state
    hipMemsetAsync(B.mkey,  0, (size_t)N * H * 4, stream);
    hipMemsetAsync(B.denom, 0, (size_t)N * H * 4, stream);
    hipMemsetAsync(B.agg,   0, (size_t)N * HC * 4, stream);

    // softmax over incoming edges + aggregation
    k_edge_max<<<cdiv((long long)EE * H, TB), TB, 0, stream>>>(src, dst, B.als, B.ald,
                                                               B.ebuf, B.mkey, E, EE, H);
    k_edge_exp<<<cdiv((long long)EE * H, TB), TB, 0, stream>>>(src, dst, B.ebuf, B.mkey,
                                                               B.denom, E, EE, H);
    k_scatter<<<cdiv((long long)EE * H * Cc, TB), TB, 0, stream>>>(src, dst, B.hbuf, B.ebuf,
                                                                   B.denom, B.agg,
                                                                   E, EE, H, Cc);
    // bias + ELU (H==1 mean is identity; H>1 is concat)
    k_finalize<<<cdiv((long long)N * HC, TB), TB, 0, stream>>>(B.agg, bias, act16_out,
                                                               N * HC, HC);
}

extern "C" void kernel_launch(void* const* d_in, const int* in_sizes, int n_in,
                              void* d_out, int out_size, void* d_ws, size_t ws_size,
                              hipStream_t stream) {
    const float* x      = (const float*)d_in[0];
    const int*   ei     = (const int*)d_in[1];
    const int*   batch  = (const int*)d_in[2];
    const float* W1     = (const float*)d_in[3];
    const float* a1s    = (const float*)d_in[4];
    const float* a1d    = (const float*)d_in[5];
    const float* b1     = (const float*)d_in[6];
    const float* W2     = (const float*)d_in[7];
    const float* a2s    = (const float*)d_in[8];
    const float* a2d    = (const float*)d_in[9];
    const float* b2     = (const float*)d_in[10];
    const float* W3     = (const float*)d_in[11];
    const float* a3s    = (const float*)d_in[12];
    const float* a3d    = (const float*)d_in[13];
    const float* b3     = (const float*)d_in[14];
    const float* fcW    = (const float*)d_in[15];
    const float* fcb    = (const float*)d_in[16];

    const int Fin = 128, Cc = 64, H1 = 8, HC1 = H1 * Cc, G = 64, O = 10;
    const int N  = in_sizes[0] / Fin;
    const int E  = in_sizes[1] / 2;
    const int EE = E + N;
    const int* src = ei;
    const int* dst = ei + E;

    // ---- carve workspace (256B-aligned bump allocator) ----
    char* p = (char*)d_ws;
    auto alloc = [&](size_t bytes) -> void* {
        void* r = (void*)p;
        p += (bytes + 255) & ~(size_t)255;
        return r;
    };
    _Float16* act16 = (_Float16*)alloc((size_t)N * HC1 * sizeof(_Float16));
    float* hbuf  = (float*)alloc((size_t)N * HC1 * 4);
    float* agg   = (float*)alloc((size_t)N * HC1 * 4);
    float* ebuf  = (float*)alloc((size_t)EE * H1 * 4);
    float* als   = (float*)alloc((size_t)N * H1 * 4);
    float* ald   = (float*)alloc((size_t)N * H1 * 4);
    unsigned* mkey = (unsigned*)alloc((size_t)N * H1 * 4);
    float* denom = (float*)alloc((size_t)N * H1 * 4);
    _Float16* W1T = (_Float16*)alloc((size_t)HC1 * Fin * 2);
    _Float16* W2T = (_Float16*)alloc((size_t)Cc * HC1 * 2);
    _Float16* W3T = (_Float16*)alloc((size_t)Cc * Cc * 2);
    float* pools = (float*)alloc((size_t)G * Cc * 4);
    float* cnts  = (float*)alloc((size_t)G * 4);

    const int TB = 256;

    // ---- weight transposes + x -> f16 ----
    k_transpose_f16<<<cdiv((long long)Fin * HC1, TB), TB, 0, stream>>>(W1, W1T, Fin, HC1);
    k_transpose_f16<<<cdiv((long long)HC1 * Cc, TB), TB, 0, stream>>>(W2, W2T, HC1, Cc);
    k_transpose_f16<<<cdiv((long long)Cc * Cc, TB), TB, 0, stream>>>(W3, W3T, Cc, Cc);
    k_cvt_f16<<<cdiv((long long)N * Fin, TB), TB, 0, stream>>>(x, act16, N * Fin);

    LayerBufs B{hbuf, agg, ebuf, als, ald, denom, mkey};

    // layer 1: 8 heads, concat
    run_gat_layer(stream, act16, W1T, Fin, H1, Cc, a1s, a1d, b1,
                  src, dst, E, EE, N, B, act16);
    // layer 2: 1 head
    run_gat_layer(stream, act16, W2T, HC1, 1, Cc, a2s, a2d, b2,
                  src, dst, E, EE, N, B, act16);
    // layer 3: 1 head
    run_gat_layer(stream, act16, W3T, Cc, 1, Cc, a3s, a3d, b3,
                  src, dst, E, EE, N, B, act16);

    // ---- global mean pool + FC head ----
    hipMemsetAsync(pools, 0, (size_t)G * Cc * 4, stream);
    hipMemsetAsync(cnts,  0, (size_t)G * 4, stream);
    k_pool<<<cdiv((long long)N * Cc, TB), TB, 0, stream>>>(agg, batch, pools, cnts, N, Cc);
    k_fc<<<cdiv((long long)G * O, TB), TB, 0, stream>>>(pools, cnts, fcW, fcb,
                                                        (float*)d_out, G, Cc, O);
}
